// Similarity_15676630630745
// MI455X (gfx1250) — matrix operand order repaired and checked by reading, not compiled, not run on hardware
//
#include <hip/hip_runtime.h>

typedef __attribute__((ext_vector_type(16))) _Float16 v16h;
typedef __attribute__((ext_vector_type(8)))  _Float16 v8h;
typedef __attribute__((ext_vector_type(8)))  float    v8f;
typedef __attribute__((ext_vector_type(4)))  float    v4f;

#define B_   16
#define NC_  1024
#define NQ_  128
#define D_   512
#define MT   64      // c-rows per workgroup
#define KT   32      // K per step (matches wmma f16 K)
#define PAD  40      // padded LDS row length in halves (80B stride, conflict-avoiding)
#define NSTEPS (D_ / KT)

static __device__ inline float dot4(v4f a, v4f b) {
  return a[0]*b[0] + a[1]*b[1] + a[2]*b[2] + a[3]*b[3];
}

static __device__ inline v8h cvt8(v4f a, v4f b) {
  v8h h;
  h[0]=(_Float16)a[0]; h[1]=(_Float16)a[1]; h[2]=(_Float16)a[2]; h[3]=(_Float16)a[3];
  h[4]=(_Float16)b[0]; h[5]=(_Float16)b[1]; h[6]=(_Float16)b[2]; h[7]=(_Float16)b[3];
  return h;
}

__global__ __launch_bounds__(256)
void sim_fused(const float* __restrict__ c,
               const float* __restrict__ q,
               const float* __restrict__ kern,
               const float* __restrict__ biasp,
               float* __restrict__ out)
{
  __shared__ __align__(16) _Float16 c_s[MT][PAD];     // (c * w_cq) in f16
  __shared__ __align__(16) _Float16 q_s[NQ_][PAD];    // raw q in f16
  __shared__ __align__(16) float cterm_s[MT];
  __shared__ __align__(16) float qterm_s[NQ_];

  const int t    = threadIdx.x;
  const int wave = t >> 5;
  const int lane = t & 31;
  const int b    = blockIdx.y;
  const int row0 = blockIdx.x * MT;

  if (t < MT)  cterm_s[t] = 0.0f;
  if (t < NQ_) qterm_s[t] = 0.0f;

  // ---- staging thread mapping ----
  const int crow = t >> 2;            // 0..63   (64 rows x 32 k, 8 elems/thread)
  const int ck   = (t & 3) * 8;       // 0,8,16,24
  const int qrow = t >> 1;            // 0..127  (128 rows x 32 k, 16 elems/thread)
  const int qk   = (t & 1) * 16;      // 0,16

  const float* cbase = c + ((size_t)b * NC_ + row0 + crow) * D_ + ck;
  const float* qbase = q + ((size_t)b * NQ_ + qrow) * D_ + qk;
  const float* wc    = kern + ck;                 // w_c  = kernel[0:D]
  const float* wq    = kern + D_ + qk;            // w_q  = kernel[D:2D]
  const float* wcq   = kern + 2 * D_ + ck;        // w_cq = kernel[2D:3D]

  float ctp = 0.0f, qtp = 0.0f;

  // ---- wave tiling: wave owns 16 rows x 64 cols (4 output tiles) ----
  const int mrow0 = (wave & 3) * 16;   // local row base within the 64-row tile
  const int n0w   = (wave >> 2) * 64;  // col base (0 or 64)
  const int g     = lane >> 4;         // lane group 0/1
  const int l16   = lane & 15;

  v8f acc[4] = {};

  // ---- software-pipeline registers: iteration k's f32 staging data ----
  v4f cx0, cx1;                 // c rows segment
  v4f qy0, qy1, qy2, qy3;       // q rows segment
  v4f wa0, wa1;                 // w_c chunk
  v4f ws0, ws1;                 // w_cq chunk
  v4f wb0, wb1, wb2, wb3;       // w_q chunk

  // prologue: prefetch k0 = 0
  {
    cx0 = *(const v4f*)(cbase);      cx1 = *(const v4f*)(cbase + 4);
    wa0 = *(const v4f*)(wc);         wa1 = *(const v4f*)(wc + 4);
    ws0 = *(const v4f*)(wcq);        ws1 = *(const v4f*)(wcq + 4);
    qy0 = *(const v4f*)(qbase);      qy1 = *(const v4f*)(qbase + 4);
    qy2 = *(const v4f*)(qbase + 8);  qy3 = *(const v4f*)(qbase + 12);
    wb0 = *(const v4f*)(wq);         wb1 = *(const v4f*)(wq + 4);
    wb2 = *(const v4f*)(wq + 8);     wb3 = *(const v4f*)(wq + 12);
  }

  for (int ks = 0; ks < NSTEPS; ++ks) {
    __syncthreads();   // previous iteration's fragment reads done before overwrite

    // ---- stage current registers into LDS (+ rank-1 term partials) ----
    ctp += dot4(cx0, wa0) + dot4(cx1, wa1);
    *(v8h*)&c_s[crow][ck] = cvt8(cx0 * ws0, cx1 * ws1);

    qtp += dot4(qy0, wb0) + dot4(qy1, wb1) + dot4(qy2, wb2) + dot4(qy3, wb3);
    *(v8h*)&q_s[qrow][qk]     = cvt8(qy0, qy1);
    *(v8h*)&q_s[qrow][qk + 8] = cvt8(qy2, qy3);

    __syncthreads();

    // ---- prefetch next iteration's global data (latency hidden by WMMA phase) ----
    if (ks + 1 < NSTEPS) {                    // wave-uniform branch: EXEC stays full
      const int kn = (ks + 1) * KT;
      cx0 = *(const v4f*)(cbase + kn);        cx1 = *(const v4f*)(cbase + kn + 4);
      wa0 = *(const v4f*)(wc + kn);           wa1 = *(const v4f*)(wc + kn + 4);
      ws0 = *(const v4f*)(wcq + kn);          ws1 = *(const v4f*)(wcq + kn + 4);
      qy0 = *(const v4f*)(qbase + kn);        qy1 = *(const v4f*)(qbase + kn + 4);
      qy2 = *(const v4f*)(qbase + kn + 8);    qy3 = *(const v4f*)(qbase + kn + 12);
      wb0 = *(const v4f*)(wq + kn);           wb1 = *(const v4f*)(wq + kn + 4);
      wb2 = *(const v4f*)(wq + kn + 8);       wb3 = *(const v4f*)(wq + kn + 12);
    }

    // ---- A fragment: row = mrow0 + l16; chunks k = g*8 and 16 + g*8 (ISA layout) ----
    v8h alo = *(const v8h*)&c_s[mrow0 + l16][g * 8];
    v8h ahi = *(const v8h*)&c_s[mrow0 + l16][16 + g * 8];
    v16h afrag = __builtin_shufflevector(alo, ahi,
        0,1,2,3,4,5,6,7,8,9,10,11,12,13,14,15);

    // ---- B fragments: column lane l16; chunks k = g*8 and 16 + g*8 (ISA layout) ----
#pragma unroll
    for (int nt = 0; nt < 4; ++nt) {
      const _Float16* bp = &q_s[n0w + nt * 16 + l16][g * 8];
      v8h blo = *(const v8h*)bp;
      v8h bhi = *(const v8h*)(bp + 16);
      v16h bfrag = __builtin_shufflevector(blo, bhi,
          0,1,2,3,4,5,6,7,8,9,10,11,12,13,14,15);
      acc[nt] = __builtin_amdgcn_wmma_f32_16x16x32_f16(
          /*neg_a=*/false, afrag, /*neg_b=*/false, bfrag,
          /*c_mod=*/(short)0, acc[nt], /*reuse_a=*/false, /*reuse_b=*/false);
    }
  }

  // ---- reduce rank-1 term partials into LDS ----
  atomicAdd(&cterm_s[crow], ctp);
  atomicAdd(&qterm_s[qrow], qtp);
  __syncthreads();

  // ---- epilogue: out = acc + c_term[m] + q_term[n] + bias ----
  const float biasv = biasp[0];
  v4f ctl = *(const v4f*)&cterm_s[mrow0 + g * 8];
  v4f cth = *(const v4f*)&cterm_s[mrow0 + g * 8 + 4];
  float ct[8] = { ctl[0], ctl[1], ctl[2], ctl[3], cth[0], cth[1], cth[2], cth[3] };

#pragma unroll
  for (int nt = 0; nt < 4; ++nt) {
    const int gcol = n0w + nt * 16 + l16;
    const float qt = qterm_s[gcol];
    float* op = out + ((size_t)b * NC_ + row0 + mrow0 + g * 8) * (size_t)NQ_ + gcol;
#pragma unroll
    for (int v = 0; v < 8; ++v) {
      op[(size_t)v * NQ_] = acc[nt][v] + ct[v] + qt + biasv;
    }
  }
}

extern "C" void kernel_launch(void* const* d_in, const int* in_sizes, int n_in,
                              void* d_out, int out_size, void* d_ws, size_t ws_size,
                              hipStream_t stream) {
  const float* c    = (const float*)d_in[0];
  const float* q    = (const float*)d_in[1];
  const float* kern = (const float*)d_in[2];
  const float* bias = (const float*)d_in[3];
  float* out        = (float*)d_out;

  dim3 grid(NC_ / MT, B_);   // (16, 16) workgroups
  sim_fused<<<grid, 256, 0, stream>>>(c, q, kern, bias, out);
}
